// Pi0ExpertStackWithPrefix_42623255446133
// MI455X (gfx1250) — compile-verified, matrix operand order checked by reading, not run on hardware
//
#include <hip/hip_runtime.h>
#include <hip/hip_bf16.h>
#include <math.h>

// ---- problem constants (from reference) ----
#define LNUM   18
#define BB     8
#define CC     50
#define AA     32
#define HH     1024
#define NHD    8     // NH
#define HDD    256
#define PP     800
#define MLPD   4096
#define T_TOT  850   // P + C
#define T_PAD  864   // padded to multiple of 32
#define MROWS  (BB*CC)   // 400, multiple of 16

typedef __attribute__((ext_vector_type(16))) __bf16 v16bf;
typedef __attribute__((ext_vector_type(8)))  __bf16 v8bf;
typedef __attribute__((ext_vector_type(4)))  __bf16 v4bf;
typedef __attribute__((ext_vector_type(8)))  float  v8f;

// ------------------------------------------------------------------
// bf16 fragment loader (row-major-over-K, 16 rows x 32 K).
// Lane layout per CDNA5 ISA 7.12.2 (16-bit A/B operands):
//   lanes 0-15 : row = lane,    K elements {0..7} then {16..23}
//   lanes 16-31: row = lane-16, K elements {8..15} then {24..31}
// Caller passes rowptr already offset by koff = (lane&16)?8:0 elements.
// Two 16-byte loads, zero conversion ALU.
// ------------------------------------------------------------------
__device__ __forceinline__ v16bf frag_ld(const __bf16* __restrict__ p) {
  v8bf lo = *reinterpret_cast<const v8bf*>(p);
  v8bf hi = *reinterpret_cast<const v8bf*>(p + 16);
  return __builtin_shufflevector(lo, hi, 0,1,2,3,4,5,6,7,8,9,10,11,12,13,14,15);
}

__device__ __forceinline__ float gelu_tanh(float x) {
  float x3 = x * x * x;
  return 0.5f * x * (1.0f + tanhf(0.7978845608028654f * (x + 0.044715f * x3)));
}

__device__ __forceinline__ float wave_red_sum(float v) {
  #pragma unroll
  for (int o = 16; o > 0; o >>= 1) v += __shfl_xor(v, o);
  return v;
}
__device__ __forceinline__ float wave_red_max(float v) {
  #pragma unroll
  for (int o = 16; o > 0; o >>= 1) v = fmaxf(v, __shfl_xor(v, o));
  return v;
}

// ------------------------------------------------------------------
// fp32 -> bf16 conversion (weight staging), 4 elems/thread
// ------------------------------------------------------------------
__global__ __launch_bounds__(256)
void cvt_bf16_kernel(const float* __restrict__ src, __bf16* __restrict__ dst, int n) {
  int i = (blockIdx.x * 256 + threadIdx.x) * 4;
  if (i >= n) return;
  float4 v = *reinterpret_cast<const float4*>(src + i);
  v4bf d;
  d[0] = (__bf16)v.x; d[1] = (__bf16)v.y; d[2] = (__bf16)v.z; d[3] = (__bf16)v.w;
  *reinterpret_cast<v4bf*>(dst + i) = d;
}

// ------------------------------------------------------------------
// Generic WMMA GEMM:  out[M,N] (epi) A[M,K] * W[N,K]^T, bf16 in, fp32 acc.
// One wave computes a 16 x (16*NT) tile: 1 A-frag reused for NT B-frags
// per K step. 8 waves / block. M mult of 16, N mult of 16*NT, K mult of 32.
// Epilogues: 0: outf = acc (+bias)        1: outf += acc
//            2: outf = gelu(acc)          3: outb = bf16(aux * acc)
// ------------------------------------------------------------------
template<int EPI, int NT>
__global__ __launch_bounds__(256)
void gemm_bf16(const __bf16* __restrict__ A, const __bf16* __restrict__ W,
               const float* __restrict__ bias, const float* __restrict__ aux,
               float* __restrict__ outf, __bf16* __restrict__ outb,
               int M, int N, int K) {
  int wave = threadIdx.x >> 5;
  int lane = threadIdx.x & 31;
  int tile = blockIdx.x * 8 + wave;
  int nGrp = N / (16 * NT);
  int total = (M >> 4) * nGrp;
  if (tile >= total) return;
  int mt = tile / nGrp;
  int ng = tile - mt * nGrp;
  int lr = lane & 15;
  int koff = (lane & 16) ? 8 : 0;
  const __bf16* aRow = A + (size_t)(mt * 16 + lr) * K + koff;
  const __bf16* wRow = W + (size_t)(ng * 16 * NT + lr) * K + koff;
  v8f acc[NT] = {};
  for (int k0 = 0; k0 < K; k0 += 32) {
    v16bf af = frag_ld(aRow + k0);
    #pragma unroll
    for (int j = 0; j < NT; ++j) {
      v16bf wf = frag_ld(wRow + (size_t)j * 16 * K + k0);
      acc[j] = __builtin_amdgcn_wmma_f32_16x16x32_bf16(false, af, false, wf,
                                                       (short)0, acc[j], false, false);
    }
  }
  int rowbase = mt * 16 + ((lane & 16) ? 8 : 0);
  #pragma unroll
  for (int j = 0; j < NT; ++j) {
    int col = ng * 16 * NT + j * 16 + lr;
    #pragma unroll
    for (int r = 0; r < 8; ++r) {
      size_t idx = (size_t)(rowbase + r) * N + col;
      float v = acc[j][r];
      if (EPI == 0)      { if (bias) v += bias[col]; outf[idx] = v; }
      else if (EPI == 1) { outf[idx] += v; }
      else if (EPI == 2) { outf[idx] = gelu_tanh(v); }
      else if (EPI == 3) { outb[idx] = (__bf16)(aux[idx] * v); }
    }
  }
}

// ------------------------------------------------------------------
// Preamble: act proj + time embedding + fused action-time MLP -> x[400,H]
// ------------------------------------------------------------------
__global__ __launch_bounds__(256)
void preamble_kernel(const float* __restrict__ noisy, const float* __restrict__ tvec,
                     const float* __restrict__ in_w, const float* __restrict__ in_b,
                     const float* __restrict__ t_in_w, const float* __restrict__ t_in_b,
                     const float* __restrict__ t_out_w, const float* __restrict__ t_out_b,
                     float* __restrict__ x) {
  __shared__ float xt[2 * HH];
  __shared__ float mid[HH];
  __shared__ float na[AA];
  int row = blockIdx.x;            // b*CC + c
  int b   = row / CC;
  int tid = threadIdx.x;
  if (tid < AA) na[tid] = noisy[(size_t)row * AA + tid];
  __syncthreads();
  for (int h = tid; h < HH; h += 256) {
    float s = in_b[h];
    const float* wr = in_w + (size_t)h * AA;
    #pragma unroll
    for (int a = 0; a < AA; ++a) s += na[a] * wr[a];
    xt[h] = s;
  }
  float tval = tvec[b];
  for (int i = tid; i < HH / 2; i += 256) {
    float frac   = (float)i / (float)(HH / 2);
    float period = 4e-3f * __expf(frac * 6.907755278982137f);  // ln(1000)
    float ang    = 6.283185307179586f * tval / period;
    float sn, cs; __sincosf(ang, &sn, &cs);
    xt[HH + i] = sn;
    xt[HH + HH / 2 + i] = cs;
  }
  __syncthreads();
  for (int h = tid; h < HH; h += 256) {
    float s = t_in_b[h];
    const float* wr = t_in_w + (size_t)h * (2 * HH);
    for (int e = 0; e < 2 * HH; ++e) s += wr[e] * xt[e];
    mid[h] = s / (1.0f + __expf(-s));  // silu
  }
  __syncthreads();
  for (int h = tid; h < HH; h += 256) {
    float s = t_out_b[h];
    const float* wr = t_out_w + (size_t)h * HH;
    for (int e = 0; e < HH; ++e) s += wr[e] * mid[e];
    x[(size_t)row * HH + h] = s;
  }
}

// ------------------------------------------------------------------
// RMSNorm, writes bf16: one block per row of [M, H]
// ------------------------------------------------------------------
__global__ __launch_bounds__(256)
void rmsnorm_kernel(const float* __restrict__ x, const float* __restrict__ w,
                    __bf16* __restrict__ out) {
  __shared__ float red[8];
  int row = blockIdx.x;
  const float* xr = x + (size_t)row * HH;
  float ss = 0.f;
  for (int i = threadIdx.x; i < HH; i += 256) { float v = xr[i]; ss += v * v; }
  ss = wave_red_sum(ss);
  if ((threadIdx.x & 31) == 0) red[threadIdx.x >> 5] = ss;
  __syncthreads();
  float tot = 0.f;
  #pragma unroll
  for (int i = 0; i < 8; ++i) tot += red[i];
  float rs = rsqrtf(tot / (float)HH + 1e-6f);
  for (int i = threadIdx.x; i < HH; i += 256)
    out[(size_t)row * HH + i] = (__bf16)(xr[i] * rs * w[i]);
}

// ------------------------------------------------------------------
// RoPE for q: reads fp32 [400, NH*HD], writes bf16 (same layout)
// ------------------------------------------------------------------
__global__ __launch_bounds__(256)
void rope_q_kernel(const float* __restrict__ in, __bf16* __restrict__ out,
                   const int* __restrict__ pos_ids) {
  int idx  = blockIdx.x * 256 + threadIdx.x;
  int ntot = MROWS * NHD * (HDD / 2);
  if (idx >= ntot) return;
  int i   = idx % (HDD / 2);
  int n   = (idx / (HDD / 2)) % NHD;
  int row = idx / ((HDD / 2) * NHD);
  int c   = row % CC;
  float pos = (float)pos_ids[c];
  float inv = __expf(-((float)i / 128.0f) * 9.210340371976184f);  // ln(10000)
  float sn, cs; __sincosf(pos * inv, &sn, &cs);
  size_t base = (size_t)row * (NHD * HDD) + (size_t)n * HDD;
  float x1 = in[base + i];
  float x2 = in[base + i + 128];
  out[base + i]       = (__bf16)(x1 * cs - x2 * sn);
  out[base + i + 128] = (__bf16)(x2 * cs + x1 * sn);
}

// RoPE for new k: fp32 in place on [400, HD]
__global__ __launch_bounds__(256)
void rope_k_kernel(float* __restrict__ x, const int* __restrict__ pos_ids) {
  int idx  = blockIdx.x * 256 + threadIdx.x;
  int ntot = MROWS * (HDD / 2);
  if (idx >= ntot) return;
  int i   = idx % (HDD / 2);
  int row = idx / (HDD / 2);
  int c   = row % CC;
  float pos = (float)pos_ids[c];
  float inv = __expf(-((float)i / 128.0f) * 9.210340371976184f);
  float sn, cs; __sincosf(pos * inv, &sn, &cs);
  size_t base = (size_t)row * HDD;
  float x1 = x[base + i];
  float x2 = x[base + i + 128];
  x[base + i]       = x1 * cs - x2 * sn;
  x[base + i + 128] = x2 * cs + x1 * sn;
}

// ------------------------------------------------------------------
// Build padded KV for one layer (bf16):
//   kk [B][T_PAD][HDD]   zero pad t>=850
//   vvT[B][HDD][T_PAD]   V transposed so PV GEMM uses the same frag loader
// ------------------------------------------------------------------
__global__ __launch_bounds__(256)
void build_kv_kernel(const float* __restrict__ pk, const float* __restrict__ pv,
                     const float* __restrict__ kn, const float* __restrict__ vn,
                     __bf16* __restrict__ kk, __bf16* __restrict__ vvT, int layer) {
  int idx  = blockIdx.x * 256 + threadIdx.x;
  int ntot = BB * T_PAD * HDD;
  if (idx >= ntot) return;
  int d = idx % HDD;
  int t = (idx / HDD) % T_PAD;
  int b = idx / (HDD * T_PAD);
  float kv = 0.f, vv = 0.f;
  if (t < PP) {
    size_t off = ((((size_t)layer * BB + b) * PP) + t) * HDD + d;
    kv = pk[off]; vv = pv[off];
  } else if (t < T_TOT) {
    size_t off = ((size_t)(b * CC + (t - PP))) * HDD + d;
    kv = kn[off]; vv = vn[off];
  }
  kk [((size_t)b * T_PAD + t) * HDD + d] = (__bf16)kv;
  vvT[((size_t)b * HDD + d) * T_PAD + t] = (__bf16)vv;
}

// ------------------------------------------------------------------
// Attention scores: S[bh][64][T_PAD] = scale * Q_bh x KK_b^T  (WMMA)
// Pad query rows (c>=50) read clamped (row 0) data; softmax discards them.
// ------------------------------------------------------------------
__global__ __launch_bounds__(256)
void attn_scores_kernel(const __bf16* __restrict__ q, const __bf16* __restrict__ kk,
                        float* __restrict__ s) {
  const int TTILES = T_PAD / 16;  // 54
  int wave = threadIdx.x >> 5;
  int lane = threadIdx.x & 31;
  int tile = blockIdx.x * 8 + wave;
  int total = BB * NHD * 4 * TTILES;
  if (tile >= total) return;
  int tt = tile % TTILES; tile /= TTILES;
  int mt = tile % 4;      tile /= 4;
  int h  = tile % NHD;
  int b  = tile / NHD;
  int lr   = lane & 15;
  int koff = (lane & 16) ? 8 : 0;
  int c    = mt * 16 + lr;
  int crow = (c < CC) ? (b * CC + c) : 0;   // clamp: pad rows discarded later
  const __bf16* aRow = q  + (size_t)crow * (NHD * HDD) + (size_t)h * HDD + koff;
  int t = tt * 16 + lr;
  const __bf16* bRow = kk + ((size_t)b * T_PAD + t) * HDD + koff;
  v8f acc = {};
  #pragma unroll
  for (int k0 = 0; k0 < HDD; k0 += 32) {
    v16bf af = frag_ld(aRow + k0);
    v16bf bf = frag_ld(bRow + k0);
    acc = __builtin_amdgcn_wmma_f32_16x16x32_bf16(false, af, false, bf,
                                                  (short)0, acc, false, false);
  }
  const float scale = 0.0625f;  // 1/sqrt(256)
  int bh = b * NHD + h;
  int col = tt * 16 + lr;
  int rowbase = mt * 16 + ((lane & 16) ? 8 : 0);
  #pragma unroll
  for (int j = 0; j < 8; ++j)
    s[((size_t)bh * 64 + rowbase + j) * T_PAD + col] = acc[j] * scale;
}

// ------------------------------------------------------------------
// Row softmax: fp32 scores -> bf16 probs. One block per (bh, row<64).
// Pad rows (c>=50) and tail cols (t>=850) are zero-filled.
// ------------------------------------------------------------------
__global__ __launch_bounds__(256)
void softmax_kernel(const float* __restrict__ s, __bf16* __restrict__ p) {
  __shared__ float redm[8], reds[8];
  int blk = blockIdx.x;            // bh*64 + r
  int r  = blk & 63;
  int bh = blk >> 6;
  const float* row = s + ((size_t)bh * 64 + r) * T_PAD;
  __bf16* prow = p + ((size_t)bh * 64 + r) * T_PAD;
  int tid = threadIdx.x;
  if (r >= CC) {
    for (int t = tid; t < T_PAD; t += 256) prow[t] = (__bf16)0.f;
    return;
  }
  float m = -1e30f;
  for (int t = tid; t < T_TOT; t += 256) m = fmaxf(m, row[t]);
  m = wave_red_max(m);
  if ((tid & 31) == 0) redm[tid >> 5] = m;
  __syncthreads();
  float mall = -1e30f;
  #pragma unroll
  for (int i = 0; i < 8; ++i) mall = fmaxf(mall, redm[i]);
  float sum = 0.f;
  for (int t = tid; t < T_TOT; t += 256) sum += __expf(row[t] - mall);
  sum = wave_red_sum(sum);
  if ((tid & 31) == 0) reds[tid >> 5] = sum;
  __syncthreads();
  float sall = 0.f;
  #pragma unroll
  for (int i = 0; i < 8; ++i) sall += reds[i];
  float inv = 1.0f / sall;
  for (int t = tid; t < T_TOT; t += 256) prow[t] = (__bf16)(__expf(row[t] - mall) * inv);
  for (int t = T_TOT + tid; t < T_PAD; t += 256) prow[t] = (__bf16)0.f;
}

// ------------------------------------------------------------------
// O = P x V (WMMA, bf16 in / bf16 out): obf[b*CC+c][h*HDD+d]
// ------------------------------------------------------------------
__global__ __launch_bounds__(256)
void attn_o_kernel(const __bf16* __restrict__ p, const __bf16* __restrict__ vvT,
                   __bf16* __restrict__ o) {
  const int NTILES = HDD / 16;  // 16
  int wave = threadIdx.x >> 5;
  int lane = threadIdx.x & 31;
  int tile = blockIdx.x * 8 + wave;
  int total = BB * NHD * 4 * NTILES;
  if (tile >= total) return;
  int nt = tile % NTILES; tile /= NTILES;
  int mt = tile % 4;      tile /= 4;
  int h  = tile % NHD;
  int b  = tile / NHD;
  int lr   = lane & 15;
  int koff = (lane & 16) ? 8 : 0;
  int bh = b * NHD + h;
  const __bf16* aRow = p   + ((size_t)bh * 64 + mt * 16 + lr) * T_PAD + koff;
  const __bf16* wRow = vvT + ((size_t)b * HDD + nt * 16 + lr) * T_PAD + koff;
  v8f acc = {};
  for (int k0 = 0; k0 < T_PAD; k0 += 32) {
    v16bf af = frag_ld(aRow + k0);
    v16bf bf = frag_ld(wRow + k0);
    acc = __builtin_amdgcn_wmma_f32_16x16x32_bf16(false, af, false, bf,
                                                  (short)0, acc, false, false);
  }
  int col     = nt * 16 + lr;
  int rowbase = ((lane & 16) ? 8 : 0);
  #pragma unroll
  for (int j = 0; j < 8; ++j) {
    int c = mt * 16 + rowbase + j;
    if (c < CC)
      o[((size_t)(b * CC + c)) * (NHD * HDD) + (size_t)h * HDD + col] = (__bf16)acc[j];
  }
}

// ------------------------------------------------------------------
// host: kernel_launch
// ------------------------------------------------------------------
extern "C" void kernel_launch(void* const* d_in, const int* in_sizes, int n_in,
                              void* d_out, int out_size, void* d_ws, size_t ws_size,
                              hipStream_t stream) {
  const float* noisy    = (const float*)d_in[0];
  const float* tstep    = (const float*)d_in[1];
  const int*   pos_ids  = (const int*)  d_in[2];
  const float* prefix_k = (const float*)d_in[3];
  const float* prefix_v = (const float*)d_in[4];
  const float* in_w     = (const float*)d_in[5];
  const float* in_b     = (const float*)d_in[6];
  const float* t_in_w   = (const float*)d_in[7];
  const float* t_in_b   = (const float*)d_in[8];
  const float* t_out_w  = (const float*)d_in[9];
  const float* t_out_b  = (const float*)d_in[10];
  const float* out_w    = (const float*)d_in[11];
  const float* out_b    = (const float*)d_in[12];
  const float* fnorm_w  = (const float*)d_in[13];
  const float* ln1_w    = (const float*)d_in[14];
  const float* q_w      = (const float*)d_in[15];
  const float* k_w      = (const float*)d_in[16];
  const float* v_w      = (const float*)d_in[17];
  const float* o_w      = (const float*)d_in[18];
  const float* ln2_w    = (const float*)d_in[19];
  const float* gate_w   = (const float*)d_in[20];
  const float* up_w     = (const float*)d_in[21];
  const float* down_w   = (const float*)d_in[22];

  // ---- workspace carve: fp32 region then bf16 region ----
  float* f32p = (float*)d_ws;
  float* xbuf = f32p;  f32p += (size_t)MROWS * HH;          // 409600
  float* qbuf = f32p;  f32p += (size_t)MROWS * NHD * HDD;   // 819200
  float* kbuf = f32p;  f32p += (size_t)MROWS * HDD;         // 102400
  float* vbuf = f32p;  f32p += (size_t)MROWS * HDD;         // 102400
  float* sbuf = f32p;  f32p += (size_t)BB * NHD * 64 * T_PAD; // 3538944
  float* gbuf = f32p;  f32p += (size_t)MROWS * MLPD;        // 1638400

  __bf16* bfp = (__bf16*)f32p;
  __bf16* hbf  = bfp;  bfp += (size_t)MROWS * HH;
  __bf16* qbf  = bfp;  bfp += (size_t)MROWS * NHD * HDD;
  __bf16* kkbf = bfp;  bfp += (size_t)BB * T_PAD * HDD;
  __bf16* vvbf = bfp;  bfp += (size_t)BB * HDD * T_PAD;
  __bf16* pbuf = bfp;  bfp += (size_t)BB * NHD * 64 * T_PAD;
  __bf16* obf  = bfp;  bfp += (size_t)MROWS * NHD * HDD;
  __bf16* mbf  = bfp;  bfp += (size_t)MROWS * MLPD;
  // per-layer staged bf16 weights (L2-resident working set, ~33.5 MB)
  __bf16* wq = bfp;  bfp += (size_t)(NHD * HDD) * HH;   // 2097152
  __bf16* wk = bfp;  bfp += (size_t)HDD * HH;           // 262144
  __bf16* wv = bfp;  bfp += (size_t)HDD * HH;           // 262144
  __bf16* wo = bfp;  bfp += (size_t)HH * (NHD * HDD);   // 2097152
  __bf16* wg = bfp;  bfp += (size_t)MLPD * HH;          // 4194304
  __bf16* wu = bfp;  bfp += (size_t)MLPD * HH;          // 4194304
  __bf16* wd = bfp;  bfp += (size_t)HH * MLPD;          // 4194304
  __bf16* wout = bfp; bfp += (size_t)AA * HH;           // 32768

  auto gblk = [](int M, int N, int NT) { return ((M / 16) * (N / (16 * NT)) + 7) / 8; };
  auto cblk = [](int n) { return (n / 4 + 255) / 256; };

  const int NQW = (NHD * HDD) * HH, NKW = HDD * HH, NOW = HH * (NHD * HDD);
  const int NGW = MLPD * HH, NDW = HH * MLPD;

  // ---- preamble + constant weight staging ----
  preamble_kernel<<<MROWS, 256, 0, stream>>>(noisy, tstep, in_w, in_b,
                                             t_in_w, t_in_b, t_out_w, t_out_b, xbuf);
  cvt_bf16_kernel<<<cblk(AA * HH), 256, 0, stream>>>(out_w, wout, AA * HH);

  const int rope_q_blk = (MROWS * NHD * (HDD / 2) + 255) / 256;
  const int rope_k_blk = (MROWS * (HDD / 2) + 255) / 256;
  const int kv_blk     = (BB * T_PAD * HDD + 255) / 256;
  const int sc_blk     = (BB * NHD * 4 * (T_PAD / 16)) / 8;   // 1728
  const int ao_blk     = (BB * NHD * 4 * (HDD / 16)) / 8;     // 512

  for (int l = 0; l < LNUM; ++l) {
    // stage this layer's weights as bf16 (stays hot in 192MB L2)
    cvt_bf16_kernel<<<cblk(NQW), 256, 0, stream>>>(q_w    + (size_t)l * NQW, wq, NQW);
    cvt_bf16_kernel<<<cblk(NKW), 256, 0, stream>>>(k_w    + (size_t)l * NKW, wk, NKW);
    cvt_bf16_kernel<<<cblk(NKW), 256, 0, stream>>>(v_w    + (size_t)l * NKW, wv, NKW);
    cvt_bf16_kernel<<<cblk(NOW), 256, 0, stream>>>(o_w    + (size_t)l * NOW, wo, NOW);
    cvt_bf16_kernel<<<cblk(NGW), 256, 0, stream>>>(gate_w + (size_t)l * NGW, wg, NGW);
    cvt_bf16_kernel<<<cblk(NGW), 256, 0, stream>>>(up_w   + (size_t)l * NGW, wu, NGW);
    cvt_bf16_kernel<<<cblk(NDW), 256, 0, stream>>>(down_w + (size_t)l * NDW, wd, NDW);

    // h = rmsnorm(x, ln1) -> bf16
    rmsnorm_kernel<<<MROWS, 256, 0, stream>>>(xbuf, ln1_w + (size_t)l * HH, hbf);
    // q,k,v projections (fp32 out for RoPE)
    gemm_bf16<0,4><<<gblk(MROWS, NHD * HDD, 4), 256, 0, stream>>>(
        hbf, wq, nullptr, nullptr, qbuf, nullptr, MROWS, NHD * HDD, HH);
    gemm_bf16<0,4><<<gblk(MROWS, HDD, 4), 256, 0, stream>>>(
        hbf, wk, nullptr, nullptr, kbuf, nullptr, MROWS, HDD, HH);
    gemm_bf16<0,4><<<gblk(MROWS, HDD, 4), 256, 0, stream>>>(
        hbf, wv, nullptr, nullptr, vbuf, nullptr, MROWS, HDD, HH);
    // RoPE: q -> bf16, new k in place
    rope_q_kernel<<<rope_q_blk, 256, 0, stream>>>(qbuf, qbf, pos_ids);
    rope_k_kernel<<<rope_k_blk, 256, 0, stream>>>(kbuf, pos_ids);
    // padded bf16 K and V^T for this layer
    build_kv_kernel<<<kv_blk, 256, 0, stream>>>(prefix_k, prefix_v, kbuf, vbuf,
                                                kkbf, vvbf, l);
    // scores (fp32) -> softmax (bf16 P) -> P*V (bf16 O)
    attn_scores_kernel<<<sc_blk, 256, 0, stream>>>(qbf, kkbf, sbuf);
    softmax_kernel<<<BB * NHD * 64, 256, 0, stream>>>(sbuf, pbuf);
    attn_o_kernel<<<ao_blk, 256, 0, stream>>>(pbuf, vvbf, obf);
    // x += o @ o_w^T
    gemm_bf16<1,4><<<gblk(MROWS, HH, 4), 256, 0, stream>>>(
        obf, wo, nullptr, nullptr, xbuf, nullptr, MROWS, HH, NHD * HDD);
    // MLP
    rmsnorm_kernel<<<MROWS, 256, 0, stream>>>(xbuf, ln2_w + (size_t)l * HH, hbf);
    gemm_bf16<2,4><<<gblk(MROWS, MLPD, 4), 256, 0, stream>>>(
        hbf, wg, nullptr, nullptr, gbuf, nullptr, MROWS, MLPD, HH);       // gelu(gate)
    gemm_bf16<3,4><<<gblk(MROWS, MLPD, 4), 256, 0, stream>>>(
        hbf, wu, nullptr, gbuf, nullptr, mbf, MROWS, MLPD, HH);           // mid=bf16(g*u)
    gemm_bf16<1,4><<<gblk(MROWS, HH, 4), 256, 0, stream>>>(
        mbf, wd, nullptr, nullptr, xbuf, nullptr, MROWS, HH, MLPD);       // x += down
  }

  // final norm + output projection (with bias) -> d_out [B,C,A] fp32
  rmsnorm_kernel<<<MROWS, 256, 0, stream>>>(xbuf, fnorm_w, hbf);
  gemm_bf16<0,2><<<gblk(MROWS, AA, 2), 256, 0, stream>>>(
      hbf, wout, out_b, nullptr, (float*)d_out, nullptr, MROWS, AA, HH);
}
